// GRUObsCell_75531294867997
// MI455X (gfx1250) — matrix operand order repaired and checked by reading, not compiled, not run on hardware
//
#include <hip/hip_runtime.h>
#include <hip/hip_bf16.h>

typedef __attribute__((ext_vector_type(16))) _Float16 v16h;
typedef __attribute__((ext_vector_type(8)))  _Float16 h8;
typedef __attribute__((ext_vector_type(4)))  _Float16 h4;
typedef __attribute__((ext_vector_type(8)))  float    v8f;
typedef __attribute__((ext_vector_type(4)))  float    f32x4;

// ---------------------------------------------------------------------------
// A-fragment (16x32 f16, M x K). Contiguous-k layout: lanes 0-15 hold row M=lane&15
// with K = kb+[0..15] (elems 0-15); lanes 16-31
// same row, K offset +16.
// ---------------------------------------------------------------------------
__device__ __forceinline__ v16h load_a_frag(const float* __restrict__ rowPtr,
                                            int kb, int laneHi) {
    const float* p = rowPtr + kb + (laneHi ? 16 : 0);
    f32x4 c0 = *(const f32x4*)(p);
    f32x4 c1 = *(const f32x4*)(p + 4);
    f32x4 c2 = *(const f32x4*)(p + 8);
    f32x4 c3 = *(const f32x4*)(p + 12);
    v16h a;
    a[0]  = (_Float16)c0.x; a[1]  = (_Float16)c0.y;
    a[2]  = (_Float16)c0.z; a[3]  = (_Float16)c0.w;
    a[4]  = (_Float16)c1.x; a[5]  = (_Float16)c1.y;
    a[6]  = (_Float16)c1.z; a[7]  = (_Float16)c1.w;
    a[8]  = (_Float16)c2.x; a[9]  = (_Float16)c2.y;
    a[10] = (_Float16)c2.z; a[11] = (_Float16)c2.w;
    a[12] = (_Float16)c3.x; a[13] = (_Float16)c3.y;
    a[14] = (_Float16)c3.z; a[15] = (_Float16)c3.w;
    return a;
}

__device__ __forceinline__ v16h load_a_frag(const _Float16* __restrict__ rowPtr,
                                            int kb, int laneHi) {
    const _Float16* p = rowPtr + kb + (laneHi ? 16 : 0);   // 16B aligned
    h8 lo = *(const h8*)(p);        // K = base .. base+7
    h8 hi = *(const h8*)(p + 8);    // K = base+8 .. base+15
    v16h a;
#pragma unroll
    for (int j = 0; j < 8; ++j) { a[j] = lo[j]; a[8 + j] = hi[j]; }
    return a;
}

// ---------------------------------------------------------------------------
// B-fragment (32x16 f16, K x N) for B = W^T: B[k][n] = W[c0+n][k].
// Lane L serves column n = L&15; elems e = 0..15 hold K = kb + e (+16 upper
// lane half) -> 16 consecutive source elements per lane.
// ---------------------------------------------------------------------------
__device__ __forceinline__ v16h load_b_frag(const float* __restrict__ wRow,
                                            int kb, int laneHi) {
    const float* p = wRow + kb + (laneHi ? 16 : 0);
    f32x4 c0 = *(const f32x4*)(p);
    f32x4 c1 = *(const f32x4*)(p + 4);
    f32x4 c2 = *(const f32x4*)(p + 8);
    f32x4 c3 = *(const f32x4*)(p + 12);
    v16h b;
    b[0]  = (_Float16)c0.x; b[1]  = (_Float16)c0.y;
    b[2]  = (_Float16)c0.z; b[3]  = (_Float16)c0.w;
    b[4]  = (_Float16)c1.x; b[5]  = (_Float16)c1.y;
    b[6]  = (_Float16)c1.z; b[7]  = (_Float16)c1.w;
    b[8]  = (_Float16)c2.x; b[9]  = (_Float16)c2.y;
    b[10] = (_Float16)c2.z; b[11] = (_Float16)c2.w;
    b[12] = (_Float16)c3.x; b[13] = (_Float16)c3.y;
    b[14] = (_Float16)c3.z; b[15] = (_Float16)c3.w;
    return b;
}

__device__ __forceinline__ v16h load_b_frag(const _Float16* __restrict__ wRow,
                                            int kb, int laneHi) {
    return *(const v16h*)(wRow + kb + (laneHi ? 16 : 0));  // 32B aligned
}

// ---------------------------------------------------------------------------
// fp32 -> fp16 prepass (grid-stride, 4 elems/thread).
// ---------------------------------------------------------------------------
__global__ __launch_bounds__(256) void cvt_f32_f16_kernel(
    const float* __restrict__ src, _Float16* __restrict__ dst, long n4) {
    long i      = (long)blockIdx.x * blockDim.x + threadIdx.x;
    long stride = (long)gridDim.x * blockDim.x;
    for (; i < n4; i += stride) {
        f32x4 v = *(const f32x4*)(src + 4 * i);
        h4 o;
        o[0] = (_Float16)v.x; o[1] = (_Float16)v.y;
        o[2] = (_Float16)v.z; o[3] = (_Float16)v.w;
        *(h4*)(dst + 4 * i) = o;
    }
}

// ---------------------------------------------------------------------------
// Gather + fp32->fp16: hs16[m, :] = (f16) h[idx[m], :]. One block per m row.
// ---------------------------------------------------------------------------
__global__ __launch_bounds__(64) void gather_cvt_h_kernel(
    const float* __restrict__ h, const int* __restrict__ idx,
    _Float16* __restrict__ hs16, int H) {
    const int m = blockIdx.x;
    const long r = (long)idx[m];
    const float*   src = h    + r * (long)H;
    _Float16*      dst = hs16 + (long)m * H;
    for (int c = 4 * threadIdx.x; c < H; c += 4 * blockDim.x) {
        f32x4 v = *(const f32x4*)(src + c);
        h4 o;
        o[0] = (_Float16)v.x; o[1] = (_Float16)v.y;
        o[2] = (_Float16)v.z; o[3] = (_Float16)v.w;
        *(h4*)(dst + c) = o;
    }
}

// ---------------------------------------------------------------------------
// Bulk copy h -> out. Non-temporal stores: out is never re-read; keep L2 for
// the h rows + staged operands the GEMM kernel wants resident.
// ---------------------------------------------------------------------------
__global__ __launch_bounds__(256) void copy_h_kernel(
    const f32x4* __restrict__ src, f32x4* __restrict__ dst, long nVec4) {
    long i      = (long)blockIdx.x * blockDim.x + threadIdx.x;
    long stride = (long)gridDim.x * blockDim.x;
    for (; i < nVec4; i += stride) {
        f32x4 v = src[i];
        __builtin_nontemporal_store(v, &dst[i]);
    }
}

// ---------------------------------------------------------------------------
// Gate epilogue for one 16x16 tile. C/D layout: lane holds column N = lane&15,
// VGPR v holds row m0 + v + 8*(lane>=16). Blends with the ORIGINAL fp32 h.
// ---------------------------------------------------------------------------
__device__ __forceinline__ void gru_epilogue(
    const v8f& air, const v8f& aiz, const v8f& ain,
    const v8f& ahr, const v8f& ahz, const v8f& ahn,
    const float* __restrict__ h, const int* __restrict__ idx,
    float* __restrict__ out, int m0, int laneHi, int col, int M, int H,
    float bir, float biz, float bin, float bhr, float bhz, float bhn) {
#pragma unroll
    for (int v = 0; v < 8; ++v) {
        int m = m0 + v + (laneHi ? 8 : 0);
        if (m >= M) continue;                   // post-WMMA predication ok
        long r = (long)idx[m];
        float hs = h[r * (long)H + col];
        float rg = 1.0f / (1.0f + __expf(-((air[v] + bir) + (ahr[v] + bhr))));
        float zg = 1.0f / (1.0f + __expf(-((aiz[v] + biz) + (ahz[v] + bhz))));
        float ng = tanhf((ain[v] + bin) + rg * (ahn[v] + bhn));
        out[r * (long)H + col] = (1.0f - zg) * ng + zg * hs;
    }
}

// ---------------------------------------------------------------------------
// GRU observed-cell update. One wave per (2 m-tiles x 1 c-tile): the three
// B fragments (gates r/z/n) feed 6 WMMAs per K-step. 8 waves / block.
// AT: A-operand source type (staged f16 or raw f32).  WT: weight type.
// When AT = _Float16, hsrc is the PRE-GATHERED hs16 (indexed by m, not idx).
// ---------------------------------------------------------------------------
template <typename WT, typename AT>
__global__ __launch_bounds__(256) void gru_obs_wmma_kernel(
    const float* __restrict__ h,     // [N, H]  (fp32, for epilogue blend)
    const AT*    __restrict__ xsrc,  // [M, Kx]
    const AT*    __restrict__ hsrc,  // f16: [M, H] gathered; f32: == h [N, H]
    const int*   __restrict__ idx,   // [M]
    const WT*    __restrict__ Wih,   // [3H, Kx]
    const WT*    __restrict__ Whh,   // [3H, H]
    const float* __restrict__ bih,   // [3H]
    const float* __restrict__ bhh,   // [3H]
    float*       __restrict__ out,   // [N, H]
    int M, int H, int Kx) {
    const int lane   = threadIdx.x & 31;
    const int laneHi = lane >> 4;
    const int lc     = lane & 15;
    // Force wave-uniform tile math into SGPRs so WMMA-containing branches are
    // scalar (s_cbranch), never EXEC-mask predication.
    const int waveId = __builtin_amdgcn_readfirstlane(threadIdx.x >> 5);

    const int nColTiles  = H >> 4;
    const int nRowTiles  = (M + 15) >> 4;
    const int nPairTiles = (nRowTiles + 1) >> 1;
    const int tile = blockIdx.x * 8 + waveId;
    if (tile >= nPairTiles * nColTiles) return;  // scalar branch

    const int mp = tile / nColTiles;
    const int ct = tile - mp * nColTiles;
    const int m0 = (mp * 2) << 4;                // first m-tile
    const int m1 = m0 + 16;                      // second m-tile
    const int c0 = ct << 4;
    const bool has2 = (m1 < M);                  // SGPR condition

    // Per-lane A source rows (clamped for tails; WMMA stays unpredicated).
    int mA0 = m0 + lc; if (mA0 >= M) mA0 = M - 1;
    int mA1 = m1 + lc; if (mA1 >= M) mA1 = M - 1;
    const bool hsGathered = (sizeof(AT) == 2);   // staged hs16 indexed by m
    const long hr0 = hsGathered ? (long)mA0 : (long)idx[mA0];
    const long hr1 = hsGathered ? (long)mA1 : (long)idx[mA1];
    const AT* __restrict__ xRow0 = xsrc + (long)mA0 * Kx;
    const AT* __restrict__ xRow1 = xsrc + (long)mA1 * Kx;
    const AT* __restrict__ hRow0 = hsrc + hr0 * H;
    const AT* __restrict__ hRow1 = hsrc + hr1 * H;

    // Weight rows for this lane's output column, per gate (r, z, n).
    const WT* __restrict__ wr_i = Wih + (long)(c0 + lc) * Kx;
    const WT* __restrict__ wz_i = Wih + (long)(H   + c0 + lc) * Kx;
    const WT* __restrict__ wn_i = Wih + (long)(2*H + c0 + lc) * Kx;
    const WT* __restrict__ wr_h = Whh + (long)(c0 + lc) * H;
    const WT* __restrict__ wz_h = Whh + (long)(H   + c0 + lc) * H;
    const WT* __restrict__ wn_h = Whh + (long)(2*H + c0 + lc) * H;

    v8f air0 = {}, aiz0 = {}, ain0 = {}, ahr0 = {}, ahz0 = {}, ahn0 = {};
    v8f air1 = {}, aiz1 = {}, ain1 = {}, ahr1 = {}, ahz1 = {}, ahn1 = {};

    // gi = x @ W_ih^T  (K = Kx)
    for (int kb = 0; kb < Kx; kb += 32) {
        v16h br = load_b_frag(wr_i, kb, laneHi);
        v16h bz = load_b_frag(wz_i, kb, laneHi);
        v16h bn = load_b_frag(wn_i, kb, laneHi);
        v16h a0 = load_a_frag(xRow0, kb, laneHi);
        air0 = __builtin_amdgcn_wmma_f32_16x16x32_f16(false, a0, false, br, (short)0, air0, false, false);
        aiz0 = __builtin_amdgcn_wmma_f32_16x16x32_f16(false, a0, false, bz, (short)0, aiz0, false, false);
        ain0 = __builtin_amdgcn_wmma_f32_16x16x32_f16(false, a0, false, bn, (short)0, ain0, false, false);
        if (has2) {
            v16h a1 = load_a_frag(xRow1, kb, laneHi);
            air1 = __builtin_amdgcn_wmma_f32_16x16x32_f16(false, a1, false, br, (short)0, air1, false, false);
            aiz1 = __builtin_amdgcn_wmma_f32_16x16x32_f16(false, a1, false, bz, (short)0, aiz1, false, false);
            ain1 = __builtin_amdgcn_wmma_f32_16x16x32_f16(false, a1, false, bn, (short)0, ain1, false, false);
        }
    }
    // gh = h[idx] @ W_hh^T  (K = H)
    for (int kb = 0; kb < H; kb += 32) {
        v16h br = load_b_frag(wr_h, kb, laneHi);
        v16h bz = load_b_frag(wz_h, kb, laneHi);
        v16h bn = load_b_frag(wn_h, kb, laneHi);
        v16h a0 = load_a_frag(hRow0, kb, laneHi);
        ahr0 = __builtin_amdgcn_wmma_f32_16x16x32_f16(false, a0, false, br, (short)0, ahr0, false, false);
        ahz0 = __builtin_amdgcn_wmma_f32_16x16x32_f16(false, a0, false, bz, (short)0, ahz0, false, false);
        ahn0 = __builtin_amdgcn_wmma_f32_16x16x32_f16(false, a0, false, bn, (short)0, ahn0, false, false);
        if (has2) {
            v16h a1 = load_a_frag(hRow1, kb, laneHi);
            ahr1 = __builtin_amdgcn_wmma_f32_16x16x32_f16(false, a1, false, br, (short)0, ahr1, false, false);
            ahz1 = __builtin_amdgcn_wmma_f32_16x16x32_f16(false, a1, false, bz, (short)0, ahz1, false, false);
            ahn1 = __builtin_amdgcn_wmma_f32_16x16x32_f16(false, a1, false, bn, (short)0, ahn1, false, false);
        }
    }

    const int   col = c0 + lc;
    const float bir = bih[col],         bhr = bhh[col];
    const float biz = bih[H + col],     bhz = bhh[H + col];
    const float bin = bih[2 * H + col], bhn = bhh[2 * H + col];

    gru_epilogue(air0, aiz0, ain0, ahr0, ahz0, ahn0,
                 h, idx, out, m0, laneHi, col, M, H, bir, biz, bin, bhr, bhz, bhn);
    if (has2)
        gru_epilogue(air1, aiz1, ain1, ahr1, ahz1, ahn1,
                     h, idx, out, m1, laneHi, col, M, H, bir, biz, bin, bhr, bhz, bhn);
}

// ---------------------------------------------------------------------------
// Launch
// ---------------------------------------------------------------------------
extern "C" void kernel_launch(void* const* d_in, const int* in_sizes, int n_in,
                              void* d_out, int out_size, void* d_ws, size_t ws_size,
                              hipStream_t stream) {
    const float* h   = (const float*)d_in[0];
    const float* X   = (const float*)d_in[1];
    const int*   idx = (const int*)d_in[2];
    const float* Wih = (const float*)d_in[3];
    const float* Whh = (const float*)d_in[4];
    const float* bih = (const float*)d_in[5];
    const float* bhh = (const float*)d_in[6];
    float*       out = (float*)d_out;

    const int  threeH = in_sizes[5];            // 3H
    const int  H      = in_sizes[4] / threeH;   // W_hh: [3H, H]
    const int  Kx     = in_sizes[3] / threeH;   // W_ih: [3H, 2*IN]
    const int  M      = in_sizes[2];
    const long NH     = (long)in_sizes[0];      // N*H

    const long nWih = (long)threeH * Kx;        // elements
    const long nWhh = (long)threeH * H;
    const long nX   = (long)M * Kx;
    const long nHs  = (long)M * H;
    const size_t needW = (size_t)(nWih + nWhh) * sizeof(_Float16);
    const size_t needA = (size_t)(nX + nHs) * sizeof(_Float16);
    const bool f16W = ws_size >= needW;
    const bool f16A = ws_size >= needW + needA;

    _Float16* wih16 = (_Float16*)d_ws;
    _Float16* whh16 = wih16 + nWih;
    _Float16* xs16  = whh16 + nWhh;
    _Float16* hs16  = xs16 + nX;

    // 1) Prepasses (all dims are multiples of 4 here).
    if (f16W) {
        long n4a = nWih >> 2, n4b = nWhh >> 2;
        cvt_f32_f16_kernel<<<dim3((unsigned)((n4a + 255) / 256)), 256, 0, stream>>>(Wih, wih16, n4a);
        cvt_f32_f16_kernel<<<dim3((unsigned)((n4b + 255) / 256)), 256, 0, stream>>>(Whh, whh16, n4b);
    }
    if (f16A) {
        long n4x = nX >> 2;
        cvt_f32_f16_kernel<<<dim3((unsigned)((n4x + 255) / 256)), 256, 0, stream>>>(X, xs16, n4x);
        gather_cvt_h_kernel<<<dim3((unsigned)M), 64, 0, stream>>>(h, idx, hs16, H);
    }

    // 2) Bulk copy h -> out (NT stores).
    long nVec4 = NH >> 2;
    long cgl   = (nVec4 + 255) / 256;
    if (cgl > 262144) cgl = 262144;
    if (cgl > 0)
        copy_h_kernel<<<dim3((unsigned)cgl), 256, 0, stream>>>(
            (const f32x4*)h, (f32x4*)out, nVec4);

    // 3) WMMA GRU update of observed rows (2 m-tiles per wave, 8 waves/block).
    const int nRowTiles = (M + 15) >> 4;
    const int nTiles    = ((nRowTiles + 1) >> 1) * (H >> 4);
    const int blocks    = (nTiles + 7) >> 3;
    if (blocks > 0) {
        if (f16A)
            gru_obs_wmma_kernel<_Float16, _Float16><<<dim3((unsigned)blocks), 256, 0, stream>>>(
                h, xs16, hs16, idx, wih16, whh16, bih, bhh, out, M, H, Kx);
        else if (f16W)
            gru_obs_wmma_kernel<_Float16, float><<<dim3((unsigned)blocks), 256, 0, stream>>>(
                h, X, h, idx, wih16, whh16, bih, bhh, out, M, H, Kx);
        else
            gru_obs_wmma_kernel<float, float><<<dim3((unsigned)blocks), 256, 0, stream>>>(
                h, X, h, idx, Wih, Whh, bih, bhh, out, M, H, Kx);
    }
}
